// RCGAN_11690900979712
// MI455X (gfx1250) — matrix operand order repaired and checked by reading, not compiled, not run on hardware
//
#include <hip/hip_runtime.h>
#include <hip/hip_bf16.h>
#include <math.h>

// ---------------- problem dims ----------------
#define B_  1024
#define T_  128
#define L_  16
#define H_  512
#define D_  64

#define BT      16          // batch rows per workgroup (one M-tile)
#define THREADS 256         // 8 waves of 32
#define NTILE_H (H_/16)     // 32 N-tiles over H
#define KCH     (H_/32)     // 16 K-chunks of 32 over H

// ---------------- workspace layout (bf16 elements) ----------------
#define OFF_WIH   0
#define OFF_WHH   (OFF_WIH + 4*H_*H_)
#define OFF_P1    (OFF_WHH + 4*H_*H_)
#define OFF_P2    (OFF_P1  + H_*H_)
#define OFF_P3    (OFF_P2  + H_*H_)
#define OFF_W2    (OFF_P3  + D_*H_)
#define OFF_W3    (OFF_W2  + H_*H_)
#define OFF_W1P   (OFF_W3  + H_*H_)          // W1 padded K 16->32
#define BF16_ELEMS (OFF_W1P + H_*32)
#define OFF_BCOMB_BYTES ((size_t)BF16_ELEMS * 2)   // f32 bih+bhh (4H)

// ---------------- LDS layout (bytes) ----------------
#define SM_CONDS 0                                   // f32 [16][512]  32768
#define SM_X     (SM_CONDS + BT*H_*4)                // bf16 [16][512] 16384
#define SM_HA    (SM_X  + BT*H_*2)                   // bf16 [16][512] 16384
#define SM_HB    (SM_HA + BT*H_*2)                   // bf16 [16][512] 16384
#define SM_Y1    (SM_HB + BT*H_*2)                   // bf16 [16][512] 16384
#define SM_Y2    (SM_Y1 + BT*H_*2)                   // bf16 [16][512] 16384
#define SM_LAB   (SM_Y2 + BT*H_*2)                   // bf16 [16][32]   1024
#define SM_ZA    (SM_LAB + BT*32*2)                  // f32 [16][512]  32768 (z stage A)
#define SM_ZB    (SM_ZA + BT*H_*4)                   // f32 [16][512]  32768 (z stage B)
#define SMEM_BYTES (SM_ZB + BT*H_*4)                 // 181248 B < 320KB WGP LDS

typedef __attribute__((ext_vector_type(16))) __bf16 v16bf;
typedef __attribute__((ext_vector_type(4)))  __bf16 v4bf;
typedef __attribute__((ext_vector_type(8)))  float  v8f;
typedef __attribute__((ext_vector_type(4)))  float  v4f;

__device__ __forceinline__ float sigm(float x) { return 1.0f / (1.0f + __expf(-x)); }

// A-matrix fragment (16x32 bf16, M rows striped over lanes; ISA 7.12.2 layout):
// lane (m = lane&15, hf = lane>>4) holds K in [8*hf, 8*hf+8) then [16+8*hf, 16+8*hf+8)
__device__ __forceinline__ v16bf load_fragA(const __bf16* base, int ld, int rr, int hf) {
    union { v16bf bf; v4f q[2]; } u;
    const __bf16* p = base + rr * ld + 8 * hf;
    u.q[0] = *(const v4f*)(p);
    u.q[1] = *(const v4f*)(p + 16);
    return u.bf;
}
// B-matrix fragment (32x16 bf16, N cols striped over lanes): weights stored row-major
// [N][K]; lane (n = lane&15, hf = lane>>4) holds K in [8*hf, 8*hf+8) then [16+8*hf, 16+8*hf+8)
__device__ __forceinline__ v16bf load_fragB(const __bf16* base, int ld, int rr, int hf) {
    union { v16bf bf; v4f q[2]; } u;
    const __bf16* p = base + rr * ld + 8 * hf;
    u.q[0] = *(const v4f*)(p);
    u.q[1] = *(const v4f*)(p + 16);
    return u.bf;
}

__device__ __forceinline__ v8f wmma_bf16(v16bf a, v16bf b, v8f c) {
    return __builtin_amdgcn_wmma_f32_16x16x32_bf16(false, a, false, b, (short)0, c, false, false);
}

// one 16x16 output tile of [16,H] @ W[N,H]^T (K = 512)
__device__ __forceinline__ v8f gemm512(const __bf16* A, const __bf16* Brow, int rr, int hf) {
    v8f acc = {0.f,0.f,0.f,0.f,0.f,0.f,0.f,0.f};
    #pragma unroll 4
    for (int kk = 0; kk < KCH; ++kk) {
        v16bf a = load_fragA(A + kk * 32, H_, rr, hf);
        v16bf b = load_fragB(Brow + kk * 32, H_, rr, hf);
        acc = wmma_bf16(a, b, acc);
    }
    return acc;
}

// async DMA of one z tile (16 rows x 512 f32 = 2048 x 16B chunks) into LDS.
// GVS addressing: mem = SGPR base + 32-bit VGPR offset; VDST = per-lane LDS byte addr.
// Tracked on ASYNCcnt; drained by s_wait_asynccnt + workgroup barrier.
__device__ __forceinline__ void issue_z_tile(unsigned lds_base, const float* gptr, int tid) {
    #pragma unroll
    for (int k = 0; k < 8; ++k) {
        int c = tid + k * THREADS;           // 16-byte chunk id
        asm volatile("global_load_async_to_lds_b128 %0, %1, %2"
                     :: "v"(lds_base + c * 16), "v"(c * 16), "s"(gptr)
                     : "memory");
    }
}
__device__ __forceinline__ void wait_async0() {
    asm volatile("s_wait_asynccnt 0x0" ::: "memory");
}

// ---------------- prep kernels ----------------
__global__ void k_f32_to_bf16(__bf16* __restrict__ dst, const float* __restrict__ src, int n) {
    int i = blockIdx.x * blockDim.x + threadIdx.x;
    int stride = gridDim.x * blockDim.x;
    for (; i < n; i += stride) dst[i] = (__bf16)src[i];
}
__global__ void k_pad_w1(__bf16* __restrict__ dst, const float* __restrict__ src) {
    int i = blockIdx.x * blockDim.x + threadIdx.x;      // H_*32
    if (i < H_ * 32) {
        int c = i & 31, rr = i >> 5;
        dst[i] = (c < L_) ? (__bf16)src[rr * L_ + c] : (__bf16)0.0f;
    }
}
__global__ void k_bias_combine(float* __restrict__ dst, const float* __restrict__ a,
                               const float* __restrict__ b, int n) {
    int i = blockIdx.x * blockDim.x + threadIdx.x;
    if (i < n) dst[i] = a[i] + b[i];
}

// ---------------- main persistent kernel ----------------
__global__ __launch_bounds__(THREADS) void rcgan_lstm(
    const float* __restrict__ labels, const float* __restrict__ z,
    const float* __restrict__ b1, const float* __restrict__ b2,
    const float* __restrict__ pb1, const float* __restrict__ pb2,
    const float* __restrict__ pb3, const float* __restrict__ bcomb,
    const __bf16* __restrict__ Wih, const __bf16* __restrict__ Whh,
    const __bf16* __restrict__ P1,  const __bf16* __restrict__ P2,
    const __bf16* __restrict__ P3,  const __bf16* __restrict__ W1p,
    const __bf16* __restrict__ W2,  const __bf16* __restrict__ W3,
    float* __restrict__ out)
{
    extern __shared__ char smem[];
    float*  conds_s = (float*)(smem + SM_CONDS);
    __bf16* x_bf    = (__bf16*)(smem + SM_X);
    __bf16* hA      = (__bf16*)(smem + SM_HA);
    __bf16* hB      = (__bf16*)(smem + SM_HB);
    __bf16* y1_bf   = (__bf16*)(smem + SM_Y1);
    __bf16* y2_bf   = (__bf16*)(smem + SM_Y2);
    __bf16* lab_bf  = (__bf16*)(smem + SM_LAB);
    float*  zA      = (float*)(smem + SM_ZA);
    float*  zB      = (float*)(smem + SM_ZB);
    // low 32 bits of a generic LDS pointer = wave-relative LDS byte address
    const unsigned zAoff = (unsigned)(size_t)zA;
    const unsigned zBoff = (unsigned)(size_t)zB;

    const int tid  = threadIdx.x;
    const int lane = tid & 31;
    const int wave = tid >> 5;
    const int rr   = lane & 15;       // A-row / B-col / D-col within tile
    const int hf   = lane >> 4;
    const int mh   = hf * 8;          // D-row block base
    const int b0   = blockIdx.x * BT;

    // kick the t=0 z-tile DMA so it overlaps the label-MLP prologue
    issue_z_tile(zAoff, z + (size_t)b0 * H_, tid);

    // ---- prologue: label MLP -> conds tile (f32, LDS-resident) ----
    for (int e = tid; e < BT * 32; e += THREADS) {
        int m = e >> 5, c = e & 31;
        lab_bf[e] = (c < L_) ? (__bf16)labels[(b0 + m) * L_ + c] : (__bf16)0.0f;
    }
    __syncthreads();

    // layer1: relu(labels @ W1^T + b1), K = 32 (padded)
    for (int jj = 0; jj < 4; ++jj) {
        int j = wave * 4 + jj;
        v16bf a = load_fragA(lab_bf, 32, rr, hf);
        v16bf b = load_fragB(W1p + (j * 16) * 32, 32, rr, hf);
        v8f acc = {0.f,0.f,0.f,0.f,0.f,0.f,0.f,0.f};
        acc = wmma_bf16(a, b, acc);
        float bias = b1[j * 16 + rr];
        for (int v = 0; v < 8; ++v)
            y1_bf[(mh + v) * H_ + j * 16 + rr] = (__bf16)fmaxf(acc[v] + bias, 0.f);
    }
    __syncthreads();
    // layer2
    for (int jj = 0; jj < 4; ++jj) {
        int j = wave * 4 + jj;
        v8f acc = gemm512(y1_bf, W2 + (size_t)(j * 16) * H_, rr, hf);
        float bias = b2[j * 16 + rr];
        for (int v = 0; v < 8; ++v)
            y2_bf[(mh + v) * H_ + j * 16 + rr] = (__bf16)fmaxf(acc[v] + bias, 0.f);
    }
    __syncthreads();
    // layer3: conds (no bias, no relu), keep f32
    for (int jj = 0; jj < 4; ++jj) {
        int j = wave * 4 + jj;
        v8f acc = gemm512(y2_bf, W3 + (size_t)(j * 16) * H_, rr, hf);
        for (int v = 0; v < 8; ++v)
            conds_s[(mh + v) * H_ + j * 16 + rr] = acc[v];
    }
    // h0 = 0
    for (int e = tid; e < BT * H_; e += THREADS) hA[e] = (__bf16)0.0f;
    __syncthreads();

    // c state lives in registers: wave owns N-tiles j = wave*4 .. wave*4+3
    float c_reg[4][8];
    #pragma unroll
    for (int jj = 0; jj < 4; ++jj)
        #pragma unroll
        for (int v = 0; v < 8; ++v) c_reg[jj][v] = 0.f;

    __bf16* h_cur = hA;  __bf16* h_nxt = hB;
    float*  z_cur = zA;  unsigned zoff_nxt = zBoff;
    float*  z_nxt = zB;  unsigned zoff_cur = zAoff;

    for (int t = 0; t < T_; ++t) {
        // drain the async z DMA issued last step (or in the prologue)
        wait_async0();
        __syncthreads();

        // stage 1: x = conds + z_t  (f32 LDS -> bf16 LDS), then kick DMA for t+1
        for (int c4 = tid; c4 < BT * H_ / 4; c4 += THREADS) {
            v4f zc = *(const v4f*)(z_cur + c4 * 4);
            v4f cc = *(const v4f*)(conds_s + c4 * 4);
            v4bf pk;
            pk[0] = (__bf16)(zc[0] + cc[0]);
            pk[1] = (__bf16)(zc[1] + cc[1]);
            pk[2] = (__bf16)(zc[2] + cc[2]);
            pk[3] = (__bf16)(zc[3] + cc[3]);
            *(v4bf*)(x_bf + c4 * 4) = pk;
        }
        if (t + 1 < T_)  // overlaps all four GEMM stages below
            issue_z_tile(zoff_nxt, z + ((size_t)(t + 1) * B_ + b0) * H_, tid);
        __syncthreads();

        // stage 2: gates = x@Wih^T + h@Whh^T + bcomb ; LSTM cell update
        for (int jj = 0; jj < 4; ++jj) {
            int j = wave * 4 + jj;
            const int rowI = j * 16;
            const __bf16* Wi = Wih + (size_t)(0 * H_ + rowI) * H_;
            const __bf16* Wf = Wih + (size_t)(1 * H_ + rowI) * H_;
            const __bf16* Wg = Wih + (size_t)(2 * H_ + rowI) * H_;
            const __bf16* Wo = Wih + (size_t)(3 * H_ + rowI) * H_;
            const __bf16* Ui = Whh + (size_t)(0 * H_ + rowI) * H_;
            const __bf16* Uf = Whh + (size_t)(1 * H_ + rowI) * H_;
            const __bf16* Ug = Whh + (size_t)(2 * H_ + rowI) * H_;
            const __bf16* Uo = Whh + (size_t)(3 * H_ + rowI) * H_;
            v8f ai = {0.f,0.f,0.f,0.f,0.f,0.f,0.f,0.f};
            v8f af = ai, ag = ai, ao = ai;
            #pragma unroll 4
            for (int kk = 0; kk < KCH; ++kk) {
                int ko = kk * 32;
                v16bf ax = load_fragA(x_bf + ko, H_, rr, hf);
                ai = wmma_bf16(ax, load_fragB(Wi + ko, H_, rr, hf), ai);
                af = wmma_bf16(ax, load_fragB(Wf + ko, H_, rr, hf), af);
                ag = wmma_bf16(ax, load_fragB(Wg + ko, H_, rr, hf), ag);
                ao = wmma_bf16(ax, load_fragB(Wo + ko, H_, rr, hf), ao);
                v16bf ah = load_fragA(h_cur + ko, H_, rr, hf);
                ai = wmma_bf16(ah, load_fragB(Ui + ko, H_, rr, hf), ai);
                af = wmma_bf16(ah, load_fragB(Uf + ko, H_, rr, hf), af);
                ag = wmma_bf16(ah, load_fragB(Ug + ko, H_, rr, hf), ag);
                ao = wmma_bf16(ah, load_fragB(Uo + ko, H_, rr, hf), ao);
            }
            const int col = rowI + rr;
            float bi = bcomb[0 * H_ + col], bf = bcomb[1 * H_ + col];
            float bg = bcomb[2 * H_ + col], bo = bcomb[3 * H_ + col];
            #pragma unroll
            for (int v = 0; v < 8; ++v) {
                float iv = sigm(ai[v] + bi);
                float fv = sigm(af[v] + bf);
                float gv = tanhf(ag[v] + bg);
                float ov = sigm(ao[v] + bo);
                float cv = fv * c_reg[jj][v] + iv * gv;
                c_reg[jj][v] = cv;
                h_nxt[(mh + v) * H_ + col] = (__bf16)(ov * tanhf(cv)); // double buffer: no WAR
            }
        }
        __syncthreads();

        // stage 3: y1 = relu(h @ P1^T + pb1)
        for (int jj = 0; jj < 4; ++jj) {
            int j = wave * 4 + jj;
            v8f acc = gemm512(h_nxt, P1 + (size_t)(j * 16) * H_, rr, hf);
            float bias = pb1[j * 16 + rr];
            for (int v = 0; v < 8; ++v)
                y1_bf[(mh + v) * H_ + j * 16 + rr] = (__bf16)fmaxf(acc[v] + bias, 0.f);
        }
        __syncthreads();

        // stage 4: y2 = relu(y1 @ P2^T + pb2)
        for (int jj = 0; jj < 4; ++jj) {
            int j = wave * 4 + jj;
            v8f acc = gemm512(y1_bf, P2 + (size_t)(j * 16) * H_, rr, hf);
            float bias = pb2[j * 16 + rr];
            for (int v = 0; v < 8; ++v)
                y2_bf[(mh + v) * H_ + j * 16 + rr] = (__bf16)fmaxf(acc[v] + bias, 0.f);
        }
        __syncthreads();

        // stage 5: out[:,t,:] = y2 @ P3^T + pb3  (D = 64 -> 4 tiles on waves 0..3)
        if (wave < 4) {
            int j = wave;
            v8f acc = gemm512(y2_bf, P3 + (size_t)(j * 16) * H_, rr, hf);
            float bias = pb3[j * 16 + rr];
            for (int v = 0; v < 8; ++v) {
                size_t o = ((size_t)(b0 + mh + v) * T_ + t) * D_ + j * 16 + rr;
                out[o] = acc[v] + bias;
            }
        }
        // swap double buffers (next step's top-of-loop wait+barrier orders stage 5)
        { __bf16* tp = h_cur; h_cur = h_nxt; h_nxt = tp; }
        { float* tp = z_cur; z_cur = z_nxt; z_nxt = tp; }
        { unsigned to = zoff_cur; zoff_cur = zoff_nxt; zoff_nxt = to; }
    }
}

// ---------------- host launcher ----------------
extern "C" void kernel_launch(void* const* d_in, const int* in_sizes, int n_in,
                              void* d_out, int out_size, void* d_ws, size_t ws_size,
                              hipStream_t stream) {
    const float* labels = (const float*)d_in[0];
    const float* z      = (const float*)d_in[1];
    const float* W1     = (const float*)d_in[2];
    const float* b1     = (const float*)d_in[3];
    const float* W2     = (const float*)d_in[4];
    const float* b2     = (const float*)d_in[5];
    const float* W3     = (const float*)d_in[6];
    const float* Wih    = (const float*)d_in[7];
    const float* Whh    = (const float*)d_in[8];
    const float* bih    = (const float*)d_in[9];
    const float* bhh    = (const float*)d_in[10];
    const float* P1     = (const float*)d_in[11];
    const float* pb1    = (const float*)d_in[12];
    const float* P2     = (const float*)d_in[13];
    const float* pb2    = (const float*)d_in[14];
    const float* P3     = (const float*)d_in[15];
    const float* pb3    = (const float*)d_in[16];
    float* out = (float*)d_out;

    __bf16* wsb = (__bf16*)d_ws;
    float* bcomb = (float*)((char*)d_ws + OFF_BCOMB_BYTES);

    // f32 -> bf16 weight conversion (L2-resident thereafter)
    k_f32_to_bf16<<<1024, 256, 0, stream>>>(wsb + OFF_WIH, Wih, 4 * H_ * H_);
    k_f32_to_bf16<<<1024, 256, 0, stream>>>(wsb + OFF_WHH, Whh, 4 * H_ * H_);
    k_f32_to_bf16<<<512,  256, 0, stream>>>(wsb + OFF_P1,  P1,  H_ * H_);
    k_f32_to_bf16<<<512,  256, 0, stream>>>(wsb + OFF_P2,  P2,  H_ * H_);
    k_f32_to_bf16<<<128,  256, 0, stream>>>(wsb + OFF_P3,  P3,  D_ * H_);
    k_f32_to_bf16<<<512,  256, 0, stream>>>(wsb + OFF_W2,  W2,  H_ * H_);
    k_f32_to_bf16<<<512,  256, 0, stream>>>(wsb + OFF_W3,  W3,  H_ * H_);
    k_pad_w1<<<(H_ * 32 + 255) / 256, 256, 0, stream>>>(wsb + OFF_W1P, W1);
    k_bias_combine<<<(4 * H_ + 255) / 256, 256, 0, stream>>>(bcomb, bih, bhh, 4 * H_);

    // persistent batch-tiled LSTM: 64 workgroups x 8 waves
    rcgan_lstm<<<B_ / BT, THREADS, SMEM_BYTES, stream>>>(
        labels, z, b1, b2, pb1, pb2, pb3, bcomb,
        wsb + OFF_WIH, wsb + OFF_WHH, wsb + OFF_P1, wsb + OFF_P2,
        wsb + OFF_P3, wsb + OFF_W1P, wsb + OFF_W2, wsb + OFF_W3, out);
}
